// SUGRL_Fast_51402168598974
// MI455X (gfx1250) — matrix operand order repaired and checked by reading, not compiled, not run on hardware
//
#include <hip/hip_runtime.h>
#include <stdint.h>

#define NNODES 50000
#define NINF   512
#define HDIM   512
#define DDIM   128
#define NADJ   3
#define NEDGE  800000
#define KEEP_SCALE 1.25f
#define DROP_THRESH 3355443u   // 0.2 * 2^24

typedef __attribute__((ext_vector_type(16))) __bf16 v16bf;
typedef __attribute__((ext_vector_type(8)))  float  v8f;
typedef __attribute__((ext_vector_type(4)))  unsigned int v4u;
typedef __attribute__((ext_vector_type(4)))  unsigned int u32x4;
typedef __attribute__((ext_vector_type(8)))  int i32x8;
typedef __attribute__((ext_vector_type(4)))  int i32x4;

union FragU { v4u q[2]; v16bf v; };

#if __has_include(<hip/amd_detail/amd_gfx1250_TDM.h>)
#define TDM_6ARG 1
#endif

static __device__ __forceinline__ uint32_t rng_hash(uint32_t x, uint32_t seed) {
    x ^= seed;
    x *= 0x9E3779B1u; x ^= x >> 16;
    x *= 0x85EBCA6Bu; x ^= x >> 13;
    x *= 0xC2B2AE35u; x ^= x >> 16;
    return x;
}

static __device__ __forceinline__ unsigned short f2bf(float f) {
    uint32_t u = __float_as_uint(f);
    uint32_t r = u + 0x7FFFu + ((u >> 16) & 1u);   // round-to-nearest-even
    return (unsigned short)(r >> 16);
}

// ---------------------------------------------------------------------------
// TDM: async 2D bf16 tile load Global -> LDS (Tensor Data Mover, D# per
// CDNA5 ISA). data_size=1 (2 bytes/elem).
// Reads with y >= tdim1 are OOB and zero-fill the LDS destination.
// ---------------------------------------------------------------------------
static __device__ __forceinline__ void tdm_load_2d_bf16(
    unsigned int lds_off_bytes, const void* gptr,
    unsigned int tdim0, unsigned int tdim1,
    unsigned int tile0, unsigned int tile1,
    unsigned long long stride0_elems) {
    unsigned long long ga = (unsigned long long)(uintptr_t)gptr;
    u32x4 g0;
    g0[0] = 1u;                                            // count=1, user D#
    g0[1] = lds_off_bytes;                                 // lds_addr [63:32]
    g0[2] = (unsigned int)(ga & 0xFFFFFFFFull);            // global_addr lo
    g0[3] = (unsigned int)((ga >> 32) & 0x01FFFFFFull)     // global_addr hi
          | (2u << 30);                                    // type=2 (image)
    i32x8 g1;
    g1[0] = (int)(1u << 16);                               // data_size=1 (2B)
    g1[1] = (int)((tdim0 & 0xFFFFu) << 16);                // tensor_dim0 lo16
    g1[2] = (int)((tdim0 >> 16) | ((tdim1 & 0xFFFFu) << 16));
    g1[3] = (int)((tdim1 >> 16) | (tile0 << 16));          // tile_dim0
    g1[4] = (int)(tile1 & 0xFFFFu);                        // tile_dim1 (dim2=0)
    g1[5] = (int)(stride0_elems & 0xFFFFFFFFull);          // dim0 stride lo
    g1[6] = (int)((stride0_elems >> 32) & 0xFFFFull);      // dim0 stride hi
    g1[7] = 0;
    i32x4 z4 = {0, 0, 0, 0};
#ifdef TDM_6ARG
    i32x8 z8 = {0, 0, 0, 0, 0, 0, 0, 0};
    __builtin_amdgcn_tensor_load_to_lds(g0, g1, z4, z4, z8, 0);
#else
    __builtin_amdgcn_tensor_load_to_lds(g0, g1, z4, z4, 0);
#endif
}

// ---------------------------------------------------------------------------
// Weight conversion: W1 [512,512] -> W1t bf16 [n][k]; W2 [512,128] -> W2t bf16
// (transposed so GEMM B-slabs are K-contiguous for the TDM)
// ---------------------------------------------------------------------------
__global__ void convert_weights_kernel(const float* __restrict__ W1,
                                       const float* __restrict__ W2,
                                       unsigned short* __restrict__ W1t,
                                       unsigned short* __restrict__ W2t) {
    int idx = blockIdx.x * blockDim.x + threadIdx.x;
    const int tot1 = HDIM * NINF;          // 262144
    if (idx < tot1) {
        int n = idx / NINF, k = idx % NINF;
        W1t[n * NINF + k] = f2bf(W1[k * HDIM + n]);
    } else {
        int j = idx - tot1;
        if (j < DDIM * HDIM) {
            int n = j / HDIM, k = j % HDIM;
            W2t[n * HDIM + k] = f2bf(W2[k * DDIM + n]);
        }
    }
}

// ---------------------------------------------------------------------------
// WMMA GEMM, 128x128 block tile, 8 waves of 32x64 each (2x4 grid of 16x16).
// Full K=512 operand slabs resident in LDS (2 x 128KB of the 320KB WGP LDS):
//   - B slab always loaded via TDM (one 16-row slice per wave)
//   - MODE 1: A slab = dropout(seq_a) f32->bf16 staged by VALU (overlaps TDM);
//             epilogue relu -> bf16 h (ws)
//   - MODE 2: A slab = h bf16 (ws) loaded via TDM, OOB rows zero-filled by
//             tensor_dim1 clamp; epilogue bias -> f32 h_a + dropout -> hp0
// ---------------------------------------------------------------------------
template <int MODE>
__global__ __launch_bounds__(256) void gemm_wmma_kernel(
    const float* __restrict__ Af32, const unsigned short* __restrict__ Abf,
    const unsigned short* __restrict__ Bt, const float* __restrict__ bias,
    unsigned short* __restrict__ outb, float* __restrict__ outf,
    float* __restrict__ outdrop) {
    constexpr int K = 512;
    constexpr int OUTW = (MODE == 1) ? HDIM : DDIM;

    extern __shared__ unsigned short smem[];
    unsigned short* Asl = smem;             // [128][512] bf16, 128KB
    unsigned short* Bsl = smem + 128 * K;   // [128][512] bf16, 128KB

    const int mblk = blockIdx.x;
    const int nblk = blockIdx.y;
    const int tid  = threadIdx.x;
    const int lane = tid & 31;
    const int w    = tid >> 5;
    const int wm   = (w & 3) * 32;    // wave M base in block tile
    const int wn   = (w >> 2) * 64;   // wave N base in block tile
    const int half = lane >> 4;       // 0/1: CDNA5 WMMA lane-half
    const int l15  = lane & 15;

    // ---- issue TDM slab loads (descriptors must be wave-uniform/SGPR) ----
    {
        int wl = __builtin_amdgcn_readfirstlane(w);
        unsigned int ldsB =
            (unsigned int)(uintptr_t)(Bsl + (size_t)wl * 16 * K);
        const unsigned short* gB = Bt + ((size_t)(nblk * 128 + wl * 16)) * K;
        tdm_load_2d_bf16(ldsB, gB, K, 16u, K, 16u, K);
        if (MODE == 2) {
            long long row0 = (long long)mblk * 128 + wl * 16;
            long long rem  = (long long)NNODES - row0;
            unsigned int td1 =
                rem <= 0 ? 0u : (rem >= 16 ? 16u : (unsigned int)rem);
            unsigned int ldsA =
                (unsigned int)(uintptr_t)(Asl + (size_t)wl * 16 * K);
            const unsigned short* gA = Abf + (size_t)row0 * K;
            tdm_load_2d_bf16(ldsA, gA, K, td1, K, 16u, K);
        }
    }

    if (MODE == 1) {
        // A slab: 128x512 elems, 256/thread, coalesced; overlaps the TDM
#pragma unroll 4
        for (int i = 0; i < 256; i++) {
            int e = tid + i * 256;
            int r = e >> 9, c = e & 511;
            int grow = mblk * 128 + r;
            float v = 0.f;
            if (grow < NNODES) {
                uint32_t gi = (uint32_t)grow * NINF + (uint32_t)c;
                v = Af32[gi];
                v = ((rng_hash(gi, 0x1234567u) & 0xFFFFFFu) >= DROP_THRESH)
                        ? v * KEEP_SCALE : 0.f;
            }
            Asl[r * K + c] = f2bf(v);
        }
    }

    __builtin_amdgcn_s_wait_tensorcnt(0);
    __syncthreads();

    v8f acc[2][4];
    for (int i = 0; i < 2; i++)
        for (int j = 0; j < 4; j++) acc[i][j] = {};

    for (int k0 = 0; k0 < K; k0 += 32) {
        // A frag (16x32 bf16): lanes 0-15 hold K {0..7,16..23}, 16-31 {8..15,24..31}
        FragU a[2], b[4];
        for (int i = 0; i < 2; i++) {
            const unsigned short* p =
                Asl + (wm + i * 16 + l15) * K + k0 + half * 8;
            a[i].q[0] = *(const v4u*)(p);
            a[i].q[1] = *(const v4u*)(p + 16);
        }
        // B frag (32x16 bf16): lane = column, lanes 0-15 K {0..7,16..23}, 16-31 {8..15,24..31}
        for (int j = 0; j < 4; j++) {
            const unsigned short* p =
                Bsl + (wn + j * 16 + l15) * K + k0 + half * 8;
            b[j].q[0] = *(const v4u*)(p);
            b[j].q[1] = *(const v4u*)(p + 16);
        }
        for (int i = 0; i < 2; i++)
            for (int j = 0; j < 4; j++)
                acc[i][j] = __builtin_amdgcn_wmma_f32_16x16x32_bf16(
                    false, a[i].v, false, b[j].v, (short)0, acc[i][j],
                    false, false);
    }

    // D layout: VGPR r -> row r (lanes 0-15) / r+8 (lanes 16-31), col = l15
    for (int i = 0; i < 2; i++) {
        for (int j = 0; j < 4; j++) {
            int colg = nblk * 128 + wn + j * 16 + l15;
            float bv = bias[colg];
            for (int r = 0; r < 8; r++) {
                int rowg = mblk * 128 + wm + i * 16 + (half ? r + 8 : r);
                if (rowg < NNODES) {
                    float v = acc[i][j][r] + bv;
                    if (MODE == 1) {
                        v = v > 0.f ? v : 0.f;
                        outb[(uint32_t)rowg * OUTW + colg] = f2bf(v);
                    } else {
                        uint32_t oi = (uint32_t)rowg * OUTW + colg;
                        outf[oi] = v;
                        outdrop[oi] =
                            ((rng_hash(oi, 0xABCDEF1u) & 0xFFFFFFu) >= DROP_THRESH)
                                ? v * KEEP_SCALE : 0.f;
                    }
                }
            }
        }
    }
}

// ---------------------------------------------------------------------------
// Zero the h_p_stack accumulation region
// ---------------------------------------------------------------------------
__global__ void zero_kernel(float* __restrict__ p, int n) {
    int i = blockIdx.x * blockDim.x + threadIdx.x;
    if (i < n) p[i] = 0.f;
}

// ---------------------------------------------------------------------------
// SpMM: one wave per edge, each lane handles 4 features (float4 gather from
// L2-resident hp0, 4 global_atomic_add_f32 scatter).
// ---------------------------------------------------------------------------
__global__ __launch_bounds__(256) void spmm_kernel(
    const float* __restrict__ vals, const int* __restrict__ rows,
    const int* __restrict__ cols, const float* __restrict__ hp0,
    float* __restrict__ outstack) {
    const int lane = threadIdx.x & 31;
    const int e = blockIdx.x * 8 + (threadIdx.x >> 5);   // E % 8 == 0
    const int a = blockIdx.y;
    const int base = a * NEDGE + e;
    const int row = rows[base];
    const int col = cols[base];
    const float val = vals[base];
    const float4 hv = ((const float4*)(hp0 + (size_t)col * DDIM))[lane];
    float* dst = outstack + (size_t)a * NNODES * DDIM + (size_t)row * DDIM + lane * 4;
    atomicAdd(dst + 0, val * hv.x);
    atomicAdd(dst + 1, val * hv.y);
    atomicAdd(dst + 2, val * hv.z);
    atomicAdd(dst + 3, val * hv.w);
}

// ---------------------------------------------------------------------------
// Fusion: mean over the A=3 stacked results
// ---------------------------------------------------------------------------
__global__ void fusion_kernel(const float* __restrict__ stack,
                              float* __restrict__ fus, int n) {
    int i = blockIdx.x * blockDim.x + threadIdx.x;
    if (i < n) {
        fus[i] = (stack[i] + stack[n + i] + stack[2 * n + i]) * (1.f / 3.f);
    }
}

// ---------------------------------------------------------------------------
extern "C" void kernel_launch(void* const* d_in, const int* in_sizes, int n_in,
                              void* d_out, int out_size, void* d_ws, size_t ws_size,
                              hipStream_t stream) {
    const float* seq_a = (const float*)d_in[0];
    const float* W1    = (const float*)d_in[1];
    const float* b1    = (const float*)d_in[2];
    const float* W2    = (const float*)d_in[3];
    const float* b2    = (const float*)d_in[4];
    const float* vals  = (const float*)d_in[5];
    const int*   rows  = (const int*)d_in[6];
    const int*   cols  = (const int*)d_in[7];

    // workspace layout
    unsigned short* W1t = (unsigned short*)d_ws;                 // 512*512
    unsigned short* W2t = W1t + HDIM * NINF;                     // 128*512
    unsigned short* hbf = W2t + DDIM * HDIM;                     // N*512 bf16
    float* hp0 = (float*)(hbf + (size_t)NNODES * HDIM);          // N*128 f32

    float* out   = (float*)d_out;
    float* ha    = out;                                          // [N,128]
    float* stack = out + (size_t)NNODES * DDIM;                  // [3,N,128]
    float* fus   = stack + (size_t)NADJ * NNODES * DDIM;         // [N,128]

    const size_t slab_bytes = 2u * 128u * 512u * sizeof(unsigned short); // 256KB

    // 1) transpose+convert weights to bf16
    convert_weights_kernel<<<(HDIM * NINF + DDIM * HDIM + 255) / 256, 256, 0,
                             stream>>>(W1, W2, W1t, W2t);

    // 2) GEMM1: h = relu(dropout(x) @ W1 + b1) -> bf16
    dim3 g1((NNODES + 127) / 128, HDIM / 128);
    gemm_wmma_kernel<1><<<g1, 256, slab_bytes, stream>>>(seq_a, nullptr, W1t,
                                                         b1, hbf, nullptr,
                                                         nullptr);

    // 3) GEMM2: h_a = h @ W2 + b2 -> f32 out; hp0 = dropout(h_a)
    dim3 g2((NNODES + 127) / 128, DDIM / 128);
    gemm_wmma_kernel<2><<<g2, 256, slab_bytes, stream>>>(nullptr, hbf, W2t,
                                                         b2, nullptr, ha, hp0);

    // 4) zero stack, SpMM per adjacency (atomics), fusion mean
    const int stack_n = NADJ * NNODES * DDIM;
    zero_kernel<<<(stack_n + 255) / 256, 256, 0, stream>>>(stack, stack_n);

    dim3 gs(NEDGE / 8, NADJ);
    spmm_kernel<<<gs, 256, 0, stream>>>(vals, rows, cols, hp0, stack);

    const int nd = NNODES * DDIM;
    fusion_kernel<<<(nd + 255) / 256, 256, 0, stream>>>(stack, fus, nd);
}
